// CifarResNet_50362786513254
// MI455X (gfx1250) — matrix operand order repaired and checked by reading, not compiled, not run on hardware
//
#include <hip/hip_runtime.h>

typedef __attribute__((ext_vector_type(16))) _Float16 v16h;
typedef __attribute__((ext_vector_type(8)))  _Float16 v8h;
typedef __attribute__((ext_vector_type(8)))  float    v8f;

#define BN_EPS 1e-5f

struct BNP { const float *g, *b, *m, *v; };

// ---------------------------------------------------------------------------
// Weight quantization (alpha = 2*mean|w|, 4-bit grid) + pack into the exact
// per-lane B-fragment layout for v_wmma_f32_16x16x32_f16:
//   lane L holds N = L%16, K = 16*(i/8) + 8*(L/16) + i%8  (i = 0..15)
// Packed index: ((coutTile*nKc + kc)*32 + lane)*16 + i
// K_global = tap*CinPad + c ; zero-filled for c >= CinReal or Kg >= Ktot.
// One workgroup per weight tensor (prep path, not perf critical).
// ---------------------------------------------------------------------------
__global__ void __launch_bounds__(256)
pack_weights(const float* __restrict__ w, _Float16* __restrict__ out,
             int Cout, int CinReal, int CinPad, int KHW, int nKc)
{
  __shared__ float red[256];
  const int tid = threadIdx.x;
  const int total = Cout * CinReal * KHW;
  float s = 0.f;
  for (int i = tid; i < total; i += 256) s += fabsf(w[i]);
  red[tid] = s;
  __syncthreads();
  for (int off = 128; off > 0; off >>= 1) {
    if (tid < off) red[tid] += red[tid + off];
    __syncthreads();
  }
  const float alpha = 2.0f * red[0] / (float)total;   // k * mean|w|, k=2
  const float sc = 7.0f / alpha;                      // 2^(4-1)-1
  const float si = alpha / 7.0f;
  const int Ktot = KHW * CinPad;
  const int packedTotal = Cout * nKc * 32;            // halves
  for (int idx = tid; idx < packedTotal; idx += 256) {
    const int i    = idx & 15;
    const int lane = (idx >> 4) & 31;
    const int kc   = (idx >> 9) % nKc;
    const int ct   = (idx >> 9) / nKc;
    const int cout = ct * 16 + (lane & 15);
    const int Kg   = kc * 32 + (i >> 3) * 16 + (lane >> 4) * 8 + (i & 7);
    float v = 0.f;
    if (Kg < Ktot) {
      const int tap = Kg / CinPad;
      const int c   = Kg - tap * CinPad;
      if (c < CinReal) {
        float wv = w[(cout * CinReal + c) * KHW + tap];   // OIHW, tap=ky*3+kx
        wv = fminf(fmaxf(wv, -alpha), alpha);
        v = rintf(wv * sc) * si;
      }
    }
    out[idx] = (_Float16)v;
  }
}

// wquant for the FC weights (kept fp32, tiny)
__global__ void __launch_bounds__(256)
quant_fc(const float* __restrict__ w, float* __restrict__ out, int total)
{
  __shared__ float red[256];
  const int tid = threadIdx.x;
  float s = 0.f;
  for (int i = tid; i < total; i += 256) s += fabsf(w[i]);
  red[tid] = s;
  __syncthreads();
  for (int off = 128; off > 0; off >>= 1) {
    if (tid < off) red[tid] += red[tid + off];
    __syncthreads();
  }
  const float alpha = 2.0f * red[0] / (float)total;
  const float sc = 7.0f / alpha, si = alpha / 7.0f;
  for (int i = tid; i < total; i += 256) {
    float v = fminf(fmaxf(w[i], -alpha), alpha);
    out[i] = rintf(v * sc) * si;
  }
}

// NCHW f32 (C=3) -> NHWC f16 with C padded to 16 (zeros)
__global__ void __launch_bounds__(256)
pad_input(const float* __restrict__ x, _Float16* __restrict__ out, int total)
{
  const int idx = blockIdx.x * 256 + threadIdx.x;
  if (idx >= total) return;
  const int c  = idx & 15;
  const int sp = idx >> 4;          // n*1024 + y*32 + x
  const int xx = sp & 31;
  const int yy = (sp >> 5) & 31;
  const int n  = sp >> 10;
  float v = 0.f;
  if (c < 3) v = x[((n * 3 + c) * 32 + yy) * 32 + xx];
  out[idx] = (_Float16)v;
}

// ---------------------------------------------------------------------------
// Implicit-GEMM conv via WMMA f16 16x16x32, fully specialized per layer shape.
// One wave = 16 output positions x (NT*16) output channels; 8 waves / block.
// NT-way output-channel register blocking: one A fragment feeds NT B fragments
// and accumulators, amortizing A address math / border predication / epilogue
// position math over NT WMMAs.  NT = 4 (Cout 64), 2 (Cout 32), 1 (Cout 16).
// K-loop fully unrolled (NKC*NT static WMMAs). With base = kc*32 + seg*16:
//   base % CIN is a multiple of 16 and 8*half < 16 <= CIN, so
//   tap = base/CIN and the Kg<KTOT guard are compile-time constants.
// Activations NHWC fp16; H/W/C all powers of two -> shifts, no divisions.
// ---------------------------------------------------------------------------
template<int HIN, int CIN, int HOUT, int COUT, int STRIDE, int PAD3,
         int HAS_RES, int DO_RELU>
__global__ void __launch_bounds__(256)
conv_wmma(const _Float16* __restrict__ act,
          const _Float16* __restrict__ wpack,
          const _Float16* __restrict__ res,     // residual, used iff HAS_RES
          _Float16* __restrict__ out,
          const float* __restrict__ bn_g, const float* __restrict__ bn_b,
          const float* __restrict__ bn_m, const float* __restrict__ bn_v,
          const float* __restrict__ alphaPtr)   // used iff DO_RELU
{
  constexpr int KTOT = PAD3 ? 9 * CIN : CIN;    // multiple of 16
  constexpr int NKC  = (KTOT + 31) / 32;
  constexpr int HW   = HOUT * HOUT;
  constexpr int NT   = (COUT >= 64) ? 4 : (COUT >= 32 ? 2 : 1);

  const int lane  = threadIdx.x & 31;
  const int wave  = threadIdx.x >> 5;
  const int tileM = blockIdx.x * 8 + wave;
  const int half  = lane >> 4;                  // 0 or 1
  const int row   = lane & 15;
  const int ctBase = blockIdx.y * NT;

  const int p  = tileM * 16 + row;
  const int n  = p / HW;                        // HW: power of two -> shift
  const int rp = p & (HW - 1);
  const int oy = rp / HOUT;
  const int ox = rp & (HOUT - 1);

  v8f acc[NT];
  {
    const v8f zero = {};
#pragma unroll
    for (int t = 0; t < NT; ++t) acc[t] = zero;
  }

#pragma unroll
  for (int kc = 0; kc < NKC; ++kc) {
    // --- A fragment (16-bit A 16x32 layout): two 8-half segments per lane ---
    v16h a;
#pragma unroll
    for (int seg = 0; seg < 2; ++seg) {
      const int base = kc * 32 + seg * 16;      // constant after unroll
      v8h av = {};
      if (base < KTOT) {                        // compile-time
        const int tap = base / CIN;             // compile-time
        const int c0  = (base % CIN) + 8 * half;
        const int dy  = PAD3 ? (tap / 3) - 1 : 0;        // compile-time
        const int dx  = PAD3 ? (tap - (tap / 3) * 3) - 1 : 0;
        const int iy  = oy * STRIDE + dy;
        const int ix  = ox * STRIDE + dx;
        if (iy >= 0 && iy < HIN && ix >= 0 && ix < HIN) {
          av = *(const v8h*)(act + (size_t)((n * HIN + iy) * HIN + ix) * CIN + c0);
        }
      }
#pragma unroll
      for (int i = 0; i < 8; ++i) a[seg * 8 + i] = av[i];
    }
    // --- NT B fragments: pre-packed, one contiguous 32B load per lane each ---
#pragma unroll
    for (int t = 0; t < NT; ++t) {
      const v16h b = *(const v16h*)(wpack +
          (((size_t)(ctBase + t) * NKC + kc) * 32 + lane) * 16);
      acc[t] = __builtin_amdgcn_wmma_f32_16x16x32_f16(
          /*neg_a=*/false, a, /*neg_b=*/false, b,
          /*c_mod=*/(short)0, acc[t], /*reuse_a=*/false, /*reuse_b=*/false);
    }
  }

  // --- Epilogue: BN + residual + clipped-ReLU quant; C/D layout M=8*half+j ---
  float inv[NT], bias[NT];
#pragma unroll
  for (int t = 0; t < NT; ++t) {
    const int ch = (ctBase + t) * 16 + row;
    inv[t]  = bn_g[ch] * rsqrtf(bn_v[ch] + BN_EPS);
    bias[t] = bn_b[ch] - bn_m[ch] * inv[t];
  }
  const float al = DO_RELU ? alphaPtr[0] : 1.0f;
  const float qs = 15.0f / al;
  const float qi = al / 15.0f;
#pragma unroll
  for (int j = 0; j < 8; ++j) {
    const int m  = 8 * half + j;
    const int pp = tileM * 16 + m;
    const int nn = pp / HW;
    const int r2 = pp & (HW - 1);
    const int yy = r2 / HOUT;
    const int xx = r2 & (HOUT - 1);
    const size_t o0 = (size_t)((nn * HOUT + yy) * HOUT + xx) * COUT
                    + (size_t)ctBase * 16 + row;
#pragma unroll
    for (int t = 0; t < NT; ++t) {
      float y = acc[t][j] * inv[t] + bias[t];
      if (HAS_RES) y += (float)res[o0 + t * 16];
      if (DO_RELU) {
        y = fminf(fmaxf(y, 0.f), al);
        y = rintf(y * qs) * qi;
      }
      out[o0 + t * 16] = (_Float16)y;
    }
  }
}

// AvgPool(8x8) + int_linear. One block (64 threads) per image.
__global__ void __launch_bounds__(64)
pool_fc(const _Float16* __restrict__ act, const float* __restrict__ fcw,
        const float* __restrict__ fcb, float* __restrict__ out)
{
  __shared__ float pooled[64];
  const int n = blockIdx.x;
  const int tid = threadIdx.x;
  const _Float16* base = act + (size_t)n * 64 * 64;   // 8*8 positions x 64 ch
  float s = 0.f;
  for (int p = 0; p < 64; ++p) s += (float)base[p * 64 + tid];
  pooled[tid] = s * (1.f / 64.f);
  __syncthreads();
  if (tid < 10) {
    float a = fcb[tid];
    for (int c = 0; c < 64; ++c) a += fcw[tid * 64 + c] * pooled[c];
    out[n * 10 + tid] = a;
  }
}

// ---------------------------------------------------------------------------
template<int HIN, int CIN, int HOUT, int COUT, int STRIDE, int PAD3,
         int HAS_RES, int DO_RELU>
static inline void launch_conv(const _Float16* act, const _Float16* wp,
                               const _Float16* res, _Float16* outp,
                               const BNP& bn, const float* alpha,
                               hipStream_t stream)
{
  constexpr int NT = (COUT >= 64) ? 4 : (COUT >= 32 ? 2 : 1);
  dim3 grid((unsigned)((1024 * HOUT * HOUT) / 128), (unsigned)(COUT / (16 * NT)));
  conv_wmma<HIN, CIN, HOUT, COUT, STRIDE, PAD3, HAS_RES, DO_RELU>
      <<<grid, 256, 0, stream>>>(act, wp, res, outp, bn.g, bn.b, bn.m, bn.v, alpha);
}

extern "C" void kernel_launch(void* const* d_in, const int* in_sizes, int n_in,
                              void* d_out, int out_size, void* d_ws, size_t ws_size,
                              hipStream_t stream)
{
  (void)in_sizes; (void)n_in; (void)out_size; (void)ws_size;
  int ci = 0;
  auto nf = [&]() { return (const float*)d_in[ci++]; };

  const float* x_in   = nf();
  const float* w_stem = nf();
  BNP bn1; bn1.g = nf(); bn1.b = nf(); bn1.m = nf(); bn1.v = nf();
  const float* alpha0 = nf();

  struct Blk {
    const float* wa; BNP bna; const float* a1;
    const float* wb; BNP bnb; const float* a2;
    const float* dw; BNP dbn;
    bool ds; int cin, cout, stride;
  } blk[9];

  static const int planes_arr[9] = {16, 16, 16, 32, 32, 32, 64, 64, 64};
  static const int stride_arr[9] = {1, 1, 1, 2, 1, 1, 2, 1, 1};
  int inplanes = 16;
  for (int b = 0; b < 9; ++b) {
    Blk& B = blk[b];
    B.cin = inplanes; B.cout = planes_arr[b]; B.stride = stride_arr[b];
    B.ds = (B.stride != 1) || (B.cin != B.cout);
    B.wa = nf(); B.bna.g = nf(); B.bna.b = nf(); B.bna.m = nf(); B.bna.v = nf();
    B.a1 = nf();
    B.wb = nf(); B.bnb.g = nf(); B.bnb.b = nf(); B.bnb.m = nf(); B.bnb.v = nf();
    B.a2 = nf();
    if (B.ds) { B.dw = nf(); B.dbn.g = nf(); B.dbn.b = nf(); B.dbn.m = nf(); B.dbn.v = nf(); }
    else       { B.dw = nullptr; B.dbn = BNP{nullptr, nullptr, nullptr, nullptr}; }
    inplanes = B.cout;
  }
  const float* fcw = nf();
  const float* fcb = nf();

  // ---- workspace carve-up (256B aligned) ----
  char* wsb = (char*)d_ws;
  size_t cur = 0;
  auto alloc = [&](size_t bytes) -> void* {
    void* p = wsb + cur;
    cur += (bytes + 255) & ~(size_t)255;
    return p;
  };
  auto packedHalves = [](int Cout, int nKc) { return (size_t)Cout * nKc * 32; };

  const int nKc_stem = 5;                           // ceil(9*16/32)
  _Float16* pk_stem = (_Float16*)alloc(packedHalves(16, nKc_stem) * 2);
  _Float16 *pkA[9], *pkB[9], *pkD[9];
  int nKcA[9], nKcB[9], nKcD[9];
  for (int b = 0; b < 9; ++b) {
    nKcA[b] = (9 * blk[b].cin + 31) / 32;
    nKcB[b] = (9 * blk[b].cout + 31) / 32;
    pkA[b] = (_Float16*)alloc(packedHalves(blk[b].cout, nKcA[b]) * 2);
    pkB[b] = (_Float16*)alloc(packedHalves(blk[b].cout, nKcB[b]) * 2);
    if (blk[b].ds) {
      nKcD[b] = (blk[b].cin + 31) / 32;
      pkD[b] = (_Float16*)alloc(packedHalves(blk[b].cout, nKcD[b]) * 2);
    } else { pkD[b] = nullptr; nKcD[b] = 0; }
  }
  float* fcq = (float*)alloc(640 * sizeof(float));
  const size_t BUF_HALVES = (size_t)1024 * 32 * 32 * 16;   // 32 MiB each
  _Float16* B0 = (_Float16*)alloc(BUF_HALVES * 2);
  _Float16* B1 = (_Float16*)alloc(BUF_HALVES * 2);
  _Float16* B2 = (_Float16*)alloc(BUF_HALVES * 2);

  // ---- weight prep ----
  pack_weights<<<1, 256, 0, stream>>>(w_stem, pk_stem, 16, 3, 16, 9, nKc_stem);
  for (int b = 0; b < 9; ++b) {
    pack_weights<<<1, 256, 0, stream>>>(blk[b].wa, pkA[b], blk[b].cout, blk[b].cin,  blk[b].cin,  9, nKcA[b]);
    pack_weights<<<1, 256, 0, stream>>>(blk[b].wb, pkB[b], blk[b].cout, blk[b].cout, blk[b].cout, 9, nKcB[b]);
    if (blk[b].ds)
      pack_weights<<<1, 256, 0, stream>>>(blk[b].dw, pkD[b], blk[b].cout, blk[b].cin, blk[b].cin, 1, nKcD[b]);
  }
  quant_fc<<<1, 256, 0, stream>>>(fcw, fcq, 640);

  // ---- input layout conversion ----
  pad_input<<<(int)(BUF_HALVES / 256), 256, 0, stream>>>(x_in, B0, (int)BUF_HALVES);

  // ---- stem: crelu(bn(conv3x3(x))) ----
  launch_conv<32, 16, 32, 16, 1, 1, 0, 1>(B0, pk_stem, nullptr, B1, bn1, alpha0, stream);

  _Float16 *curB = B1, *t1 = B0, *t2 = B2;

  // stage 1: blocks 0..2  (16->16, H=32)
  for (int b = 0; b < 3; ++b) {
    launch_conv<32, 16, 32, 16, 1, 1, 0, 1>(curB, pkA[b], nullptr, t1, blk[b].bna, blk[b].a1, stream);
    launch_conv<32, 16, 32, 16, 1, 1, 1, 1>(t1, pkB[b], curB, t2, blk[b].bnb, blk[b].a2, stream);
    _Float16* old = curB; curB = t2; t2 = old;
  }
  // block 3: 16->32, stride 2, downsample
  launch_conv<32, 16, 16, 32, 2, 1, 0, 1>(curB, pkA[3], nullptr, t1, blk[3].bna, blk[3].a1, stream);
  launch_conv<32, 16, 16, 32, 2, 0, 0, 0>(curB, pkD[3], nullptr, t2, blk[3].dbn, nullptr, stream);
  launch_conv<16, 32, 16, 32, 1, 1, 1, 1>(t1, pkB[3], t2, curB, blk[3].bnb, blk[3].a2, stream);
  // stage 2: blocks 4..5  (32->32, H=16)
  for (int b = 4; b < 6; ++b) {
    launch_conv<16, 32, 16, 32, 1, 1, 0, 1>(curB, pkA[b], nullptr, t1, blk[b].bna, blk[b].a1, stream);
    launch_conv<16, 32, 16, 32, 1, 1, 1, 1>(t1, pkB[b], curB, t2, blk[b].bnb, blk[b].a2, stream);
    _Float16* old = curB; curB = t2; t2 = old;
  }
  // block 6: 32->64, stride 2, downsample
  launch_conv<16, 32, 8, 64, 2, 1, 0, 1>(curB, pkA[6], nullptr, t1, blk[6].bna, blk[6].a1, stream);
  launch_conv<16, 32, 8, 64, 2, 0, 0, 0>(curB, pkD[6], nullptr, t2, blk[6].dbn, nullptr, stream);
  launch_conv<8, 64, 8, 64, 1, 1, 1, 1>(t1, pkB[6], t2, curB, blk[6].bnb, blk[6].a2, stream);
  // stage 3: blocks 7..8  (64->64, H=8)
  for (int b = 7; b < 9; ++b) {
    launch_conv<8, 64, 8, 64, 1, 1, 0, 1>(curB, pkA[b], nullptr, t1, blk[b].bna, blk[b].a1, stream);
    launch_conv<8, 64, 8, 64, 1, 1, 1, 1>(t1, pkB[b], curB, t2, blk[b].bnb, blk[b].a2, stream);
    _Float16* old = curB; curB = t2; t2 = old;
  }

  // AvgPool(8) + FC(64->10)
  pool_fc<<<1024, 64, 0, stream>>>(curB, fcq, fcb, (float*)d_out);
}
